// MultiheadSelfAttention_39599598469771
// MI455X (gfx1250) — matrix operand order repaired and checked by reading, not compiled, not run on hardware
//
#include <hip/hip_runtime.h>
#include <hip/hip_bf16.h>

// MI455X (gfx1250) multihead self-attention with RoPE.
// f16 WMMA (16x16x32, fp32 accumulate) for all four GEMMs + flash attention.
// Flash attention stages K/V tiles into LDS with CDNA5 async copies
// (global_load_async_to_lds_b128 / ASYNCcnt), double-buffered, shared by
// 4 waves per block.

#define BATCH   4
#define SEQ     2048
#define DMODEL  1024
#define NHEADS  16
#define HDIM    64
#define MROWS   (BATCH * SEQ)      // 8192
#define LN_THETA 9.210340371976184f // ln(10000)

typedef __attribute__((ext_vector_type(16))) _Float16 v16h;
typedef __attribute__((ext_vector_type(8)))  _Float16 v8h;
typedef __attribute__((ext_vector_type(4)))  _Float16 v4h;
typedef __attribute__((ext_vector_type(8)))  float    v8f;

// ---------------------------------------------------------------------------
// WMMA helpers (CDNA5 layouts, wave32)
// A 16x32 f16: lane = 16*g + m ; element i holds K = i + 16*g (k order chosen here, not the ISA table order)
// B 32x16 f16: lane = 16*g + n ; element i holds K = i + 16*g (same chosen order as A, not the ISA table order)
// C/D 16x16 f32: lane = 16*g + n ; VGPR r holds (M = r + 8*g, N = n)
// ---------------------------------------------------------------------------
static __device__ __forceinline__ v8f wmma_f16(v16h a, v16h b, v8f c) {
  return __builtin_amdgcn_wmma_f32_16x16x32_f16(
      /*neg_a=*/false, a, /*neg_b=*/false, b,
      /*c_mod=*/(short)0, c, /*reuse_a=*/false, /*reuse_b=*/false);
}

static __device__ __forceinline__ v16h load_a_g(const _Float16* __restrict__ p0, int ld) {
  const int lane = threadIdx.x & 31;
  const int g = lane >> 4, m = lane & 15;
  const _Float16* p = p0 + (size_t)m * ld + 16 * g;
  return *(const v16h*)p;
}

static __device__ __forceinline__ v16h load_b_g(const _Float16* __restrict__ p0, int ld) {
  const int lane = threadIdx.x & 31;
  const int g = lane >> 4, n = lane & 15;
  return *(const v16h*)(p0 + (size_t)n * ld + 16 * g);
}

// CDNA5 async copy: LDS[lds_off..+15] = MEM[gaddr..+15], tracked by ASYNCcnt.
static __device__ __forceinline__ void async_g2l_b128(unsigned lds_off, const void* gaddr) {
  asm volatile("global_load_async_to_lds_b128 %0, %1, off"
               :: "v"(lds_off), "v"(gaddr) : "memory");
}

#define COMPILER_FENCE() asm volatile("" ::: "memory")

// ---------------------------------------------------------------------------
// f32 -> f16 convert (4 elements/thread, 16B loads / 8B stores)
// ---------------------------------------------------------------------------
__global__ void cvt_f32_f16_kernel(const float* __restrict__ s,
                                   _Float16* __restrict__ d, int n) {
  int i = (blockIdx.x * blockDim.x + threadIdx.x) * 4;
  if (i < n) {
    float4 v = *(const float4*)(s + i);
    v4h h;
    h[0] = (_Float16)v.x; h[1] = (_Float16)v.y;
    h[2] = (_Float16)v.z; h[3] = (_Float16)v.w;
    *(v4h*)(d + i) = h;
  }
}

// ---------------------------------------------------------------------------
// Y[M,N] = X[M,K] * W[N,K]^T   (einsum 'mk,nk->mn'), f16 in, fp32 accum.
// Per wave: 32x64 tile (2x4 accumulators), block = 8 waves -> 64x256 tile.
// mode 0: store f32 row-major [M,N]  (final out-projection)
// mode 1: fused RoPE, store f16 at [b,h,s,hd]   (Q and K)
// mode 2: store f16 transposed at [b,h,hd,s]    (V, so P*V B-frags are dense)
// ---------------------------------------------------------------------------
__global__ __launch_bounds__(256)
void gemm_xwT_kernel(const _Float16* __restrict__ X,
                     const _Float16* __restrict__ W,
                     void* __restrict__ out,
                     const int* __restrict__ tokpos,
                     int Ktot, int Ntot, int mode) {
  const int lane = threadIdx.x & 31;
  const int wave = threadIdx.x >> 5;
  const int g = lane >> 4, n = lane & 15;
  const int m0 = blockIdx.x * 64 + (wave >> 2) * 32;
  const int n0 = blockIdx.y * 256 + (wave & 3) * 64;

  v8f acc[2][4] = {};

  for (int kb = 0; kb < Ktot; kb += 32) {
    // Prefetch the next K-slab of the activation stream (global_prefetch_b8).
    __builtin_prefetch(X + (size_t)m0 * Ktot + kb + 64, 0, 1);
    v16h a0 = load_a_g(X + (size_t)m0 * Ktot + kb, Ktot);
    v16h a1 = load_a_g(X + (size_t)(m0 + 16) * Ktot + kb, Ktot);
#pragma unroll
    for (int j = 0; j < 4; ++j) {
      v16h bj = load_b_g(W + (size_t)(n0 + 16 * j) * Ktot + kb, Ktot);
      acc[0][j] = wmma_f16(a0, bj, acc[0][j]);
      acc[1][j] = wmma_f16(a1, bj, acc[1][j]);
    }
  }

#pragma unroll
  for (int i = 0; i < 2; ++i) {
#pragma unroll
    for (int j = 0; j < 4; ++j) {
#pragma unroll
      for (int e = 0; e < 8; ++e) {
        const int row = m0 + 16 * i + e + 8 * g;  // flat (b*SEQ + s)
        const int col = n0 + 16 * j + n;          // output feature
        float val = acc[i][j][e];
        if (mode == 0) {
          ((float*)out)[(size_t)row * Ntot + col] = val;
        } else if (mode == 1) {
          // RoPE: pair (even,odd) feature lives in adjacent lanes of C-layout.
          float partner = __shfl_xor(val, 1, 32);
          const int bb = row >> 11, s = row & (SEQ - 1);
          const int hh = col >> 6,  d = col & (HDIM - 1);
          float pos = (float)tokpos[s];
          float inv = __expf(-(float)(d >> 1) * (2.0f / HDIM) * LN_THETA);
          float ang = pos * inv;
          float sn, cs;
          __sincosf(ang, &sn, &cs);
          float res = (d & 1) ? fmaf(partner, sn, val * cs)   // odd:  x1*sin + x2*cos
                              : fmaf(val, cs, -partner * sn); // even: x1*cos - x2*sin
          ((_Float16*)out)[(((size_t)(bb * NHEADS + hh) * SEQ) + s) * HDIM + d] =
              (_Float16)res;
        } else {
          const int bb = row >> 11, s = row & (SEQ - 1);
          const int hh = col >> 6,  d = col & (HDIM - 1);
          ((_Float16*)out)[(((size_t)(bb * NHEADS + hh) * HDIM) + d) * SEQ + s] =
              (_Float16)val;
        }
      }
    }
  }
}

// ---------------------------------------------------------------------------
// Flash attention. Block = 4 waves = 64 query rows of one (b,h); each wave
// owns 16 rows. K/V 32-key tiles staged into LDS with async copies
// (double-buffered) and shared by all 4 waves. Causal.
// q,k: f16 [b,h,s,hd] ; vt: f16 [b,h,hd,s] ; att: f16 merged [b,s,h*hd].
// ---------------------------------------------------------------------------
__global__ __launch_bounds__(128)
void flash_attn_kernel(const _Float16* __restrict__ q,
                       const _Float16* __restrict__ kk,
                       const _Float16* __restrict__ vt,
                       _Float16* __restrict__ att) {
  __shared__ _Float16 Kl[2][32 * HDIM];   // [buf][key][hd]       4KB each
  __shared__ _Float16 Vl[2][HDIM * 32];   // [buf][hd][key]       4KB each
  __shared__ _Float16 plds[4][16 * 32];   // per-wave P transpose staging

  const int t    = threadIdx.x;
  const int lane = t & 31;
  const int wave = t >> 5;
  const int g = lane >> 4, nn = lane & 15;
  const int bh = blockIdx.x;                // b*NHEADS + h
  const int b = bh >> 4, h = bh & 15;
  const int q0  = blockIdx.y * 64;          // block's first query row
  const int q0w = q0 + 16 * wave;           // this wave's first query row

  const _Float16* qh = q  + (size_t)bh * SEQ * HDIM;
  const _Float16* kh = kk + (size_t)bh * SEQ * HDIM;
  const _Float16* vh = vt + (size_t)bh * HDIM * SEQ;

  const unsigned kBase = (unsigned)(uintptr_t)&Kl[0][0];
  const unsigned vBase = (unsigned)(uintptr_t)&Vl[0][0];

  // Stage one 32-key K tile (contiguous 4KB) + V tile (64 rows x 64B).
  auto stage = [&](int kb, int ib) {
    {
      const char* gp = (const char*)(kh + (size_t)kb * HDIM) + t * 32;
      unsigned lp = kBase + (unsigned)(ib * 4096 + t * 32);
      async_g2l_b128(lp, gp);
      async_g2l_b128(lp + 16, gp + 16);
    }
    {
      const int r = t >> 1, hf = t & 1;     // V^T row (hd), 32B half
      const char* gp = (const char*)(vh + (size_t)r * SEQ + kb) + hf * 32;
      unsigned lp = vBase + (unsigned)(ib * 4096 + r * 64 + hf * 32);
      async_g2l_b128(lp, gp);
      async_g2l_b128(lp + 16, gp + 16);
    }
  };

  // Q tile (16 x 64) as two A fragments, resident for the whole loop.
  v16h qa0 = load_a_g(qh + (size_t)q0w * HDIM +  0, HDIM);
  v16h qa1 = load_a_g(qh + (size_t)q0w * HDIM + 32, HDIM);

  v8f o[4] = {};
  float mrow[8], lrow[8];
#pragma unroll
  for (int e = 0; e < 8; ++e) { mrow[e] = -1e30f; lrow[e] = 0.0f; }

  const int kbmax = q0 + 63;                // last key row the block needs
  stage(0, 0);
  int ib = 0;
  for (int kb = 0; kb <= kbmax; kb += 32, ib ^= 1) {
    const bool hasNext = (kb + 32 <= kbmax);
    if (hasNext) {
      stage(kb + 32, ib ^ 1);               // prefetch next tile (async)
      asm volatile("s_wait_asynccnt 0x4" ::: "memory");  // own prev 4 done
    } else {
      asm volatile("s_wait_asynccnt 0x0" ::: "memory");
    }
    __syncthreads();                        // all waves' tiles visible

    if (kb <= q0w + 15) {                   // causal: this wave has live keys
      // K B-fragments from LDS: kf[sub][ks] covers keys 16*sub.., hd 32*ks..
      v16h kf[2][2];
#pragma unroll
      for (int sub = 0; sub < 2; ++sub) {
#pragma unroll
        for (int ks = 0; ks < 2; ++ks) {
#pragma unroll
          for (int i = 0; i < 16; ++i)
            kf[sub][ks][i] = Kl[ib][(16 * sub + nn) * HDIM + 32 * ks + 16 * g + i];
        }
      }
      v8f zc = {};
      v8f s0 = wmma_f16(qa0, kf[0][0], zc); s0 = wmma_f16(qa1, kf[0][1], s0);
      v8f s1 = wmma_f16(qa0, kf[1][0], zc); s1 = wmma_f16(qa1, kf[1][1], s1);

      float p0[8], p1[8];
#pragma unroll
      for (int e = 0; e < 8; ++e) {
        const int row = q0w + e + 8 * g;
        float x0 = s0[e] * 0.125f;                 // 1/sqrt(64)
        float x1 = s1[e] * 0.125f;
        if (kb + nn      > row) x0 = -1e30f;       // causal mask
        if (kb + 16 + nn > row) x1 = -1e30f;
        float tm = fmaxf(x0, x1);                  // row-max over 16 lanes
        tm = fmaxf(tm, __shfl_xor(tm, 1, 32));
        tm = fmaxf(tm, __shfl_xor(tm, 2, 32));
        tm = fmaxf(tm, __shfl_xor(tm, 4, 32));
        tm = fmaxf(tm, __shfl_xor(tm, 8, 32));
        float mnew = fmaxf(mrow[e], tm);
        float alpha = __expf(mrow[e] - mnew);
        mrow[e] = mnew;
        x0 = __expf(x0 - mnew);
        x1 = __expf(x1 - mnew);
        float ts = x0 + x1;
        ts += __shfl_xor(ts, 1, 32);
        ts += __shfl_xor(ts, 2, 32);
        ts += __shfl_xor(ts, 4, 32);
        ts += __shfl_xor(ts, 8, 32);
        lrow[e] = lrow[e] * alpha + ts;
        o[0][e] *= alpha; o[1][e] *= alpha; o[2][e] *= alpha; o[3][e] *= alpha;
        p0[e] = x0; p1[e] = x1;
      }

      // Transpose P (accum layout -> A layout) through this wave's LDS slab.
      // Same-wave LDS ops retire in order; fences stop compiler reordering.
      COMPILER_FENCE();
#pragma unroll
      for (int e = 0; e < 8; ++e) {
        const int rr = e + 8 * g;
        plds[wave][rr * 32 + nn]      = (_Float16)p0[e];
        plds[wave][rr * 32 + 16 + nn] = (_Float16)p1[e];
      }
      COMPILER_FENCE();
      v16h pa;
#pragma unroll
      for (int i = 0; i < 8; ++i) {
        pa[i]     = plds[wave][nn * 32 + 16 * g + i];
        pa[8 + i] = plds[wave][nn * 32 + 16 * g + 8 + i];
      }
      COMPILER_FENCE();

      // O += P * V from the LDS V^T tile.
#pragma unroll
      for (int j = 0; j < 4; ++j) {
        v16h bv;
#pragma unroll
        for (int i = 0; i < 16; ++i)
          bv[i] = Vl[ib][(16 * j + nn) * 32 + 16 * g + i];
        o[j] = wmma_f16(pa, bv, o[j]);
      }
    }

    __syncthreads();                        // done reading before overwrite
  }

  // Normalize and scatter into merged [b, s, h*hd] layout.
#pragma unroll
  for (int j = 0; j < 4; ++j) {
#pragma unroll
    for (int e = 0; e < 8; ++e) {
      const int s = q0w + e + 8 * g;
      const int d = 16 * j + nn;
      att[((size_t)b * SEQ + s) * DMODEL + h * HDIM + d] =
          (_Float16)(o[j][e] / lrow[e]);
    }
  }
}

// ---------------------------------------------------------------------------
// Host-side launcher
// ---------------------------------------------------------------------------
extern "C" void kernel_launch(void* const* d_in, const int* in_sizes, int n_in,
                              void* d_out, int out_size, void* d_ws, size_t ws_size,
                              hipStream_t stream) {
  (void)in_sizes; (void)n_in; (void)out_size; (void)ws_size;
  const float* x  = (const float*)d_in[0];
  const int*   tp = (const int*)d_in[1];
  const float* wq = (const float*)d_in[2];
  const float* wk = (const float*)d_in[3];
  const float* wv = (const float*)d_in[4];
  const float* wo = (const float*)d_in[5];
  float* out = (float*)d_out;

  // Workspace carve-up (f16 elements). Att aliases Xh: Xh is dead once the
  // V projection has been consumed, Att is written strictly afterwards.
  _Float16* f16ws = (_Float16*)d_ws;
  const size_t NX = (size_t)MROWS * DMODEL;   // 8M elems
  const size_t NW = (size_t)DMODEL * DMODEL;  // 1M elems
  _Float16* Xh  = f16ws;
  _Float16* Wqh = Xh  + NX;
  _Float16* Wkh = Wqh + NW;
  _Float16* Wvh = Wkh + NW;
  _Float16* Woh = Wvh + NW;
  _Float16* Qf  = Woh + NW;
  _Float16* Kf  = Qf  + NX;
  _Float16* Vt  = Kf  + NX;
  _Float16* Att = Xh;  // alias (see above)

  // 1) f32 -> f16
  cvt_f32_f16_kernel<<<(int)(NX / 1024), 256, 0, stream>>>(x,  Xh,  (int)NX);
  cvt_f32_f16_kernel<<<(int)(NW / 1024), 256, 0, stream>>>(wq, Wqh, (int)NW);
  cvt_f32_f16_kernel<<<(int)(NW / 1024), 256, 0, stream>>>(wk, Wkh, (int)NW);
  cvt_f32_f16_kernel<<<(int)(NW / 1024), 256, 0, stream>>>(wv, Wvh, (int)NW);
  cvt_f32_f16_kernel<<<(int)(NW / 1024), 256, 0, stream>>>(wo, Woh, (int)NW);

  // 2) Q/K/V projections (fused RoPE for Q,K; transposed store for V)
  dim3 gg(MROWS / 64, DMODEL / 256);
  gemm_xwT_kernel<<<gg, 256, 0, stream>>>(Xh, Wqh, (void*)Qf, tp, DMODEL, DMODEL, 1);
  gemm_xwT_kernel<<<gg, 256, 0, stream>>>(Xh, Wkh, (void*)Kf, tp, DMODEL, DMODEL, 1);
  gemm_xwT_kernel<<<gg, 256, 0, stream>>>(Xh, Wvh, (void*)Vt, tp, DMODEL, DMODEL, 2);

  // 3) Causal flash attention (async LDS staging, 4 waves/block)
  flash_attn_kernel<<<dim3(BATCH * NHEADS, SEQ / 64), 128, 0, stream>>>(Qf, Kf, Vt, Att);

  // 4) Output projection -> fp32 result
  gemm_xwT_kernel<<<gg, 256, 0, stream>>>(Att, Woh, (void*)out, nullptr,
                                          DMODEL, DMODEL, 0);
}
